// DenseContrastiveHead_42314017800611
// MI455X (gfx1250) — matrix operand order repaired and checked by reading, not compiled, not run on hardware
//
#include <hip/hip_runtime.h>
#include <hip/hip_bf16.h>

typedef __attribute__((ext_vector_type(16))) __bf16 v16bf;
typedef __attribute__((ext_vector_type(8)))  __bf16 v8bf;
typedef __attribute__((ext_vector_type(8)))  float  v8f;

#define SP   1024   // h*w
#define DD   256    // feature dim of 'of'
#define CC   128    // feature dim of x / ag / ad
#define BB   64     // batch per side
#define TEMP_INV 10.0f

// ---- workspace layout (bytes) ----
static constexpr size_t OFNT_OFF  = 0;                                   // __bf16[128*1024*256]  normalized, spatial-major
static constexpr size_t XINV_OFF  = (size_t)128 * SP * DD * 2;           // float[128*1024]
static constexpr size_t IND_OFF   = XINV_OFF + (size_t)128 * SP * 4;     // int[2*64*1024]
static constexpr size_t AGN_OFF   = IND_OFF  + (size_t)2 * BB * SP * 4;  // float[128*128]
static constexpr size_t DPART_OFF = AGN_OFF  + (size_t)128 * CC * 4;     // float[512]
static constexpr size_t GSUM_OFF  = DPART_OFF + 512 * 4;                 // float[1]

// ------------------------------------------------------------------
// 1) normalize 'of' columns (over d) and store bf16, spatial-major:
//    ofnT[b][s][d] = of[b][d][s] / max(||of[b][:,s]||, 1e-12)
// ------------------------------------------------------------------
__global__ void prep_ofnT(const float* __restrict__ ofq,
                          const float* __restrict__ ofk,
                          __bf16* __restrict__ ofnT) {
    int b0 = blockIdx.y;
    int s  = blockIdx.x * 256 + threadIdx.x;
    const float* src = (b0 < BB) ? (ofq + (size_t)b0 * DD * SP)
                                 : (ofk + (size_t)(b0 - BB) * DD * SP);
    float ss = 0.f;
    for (int d = 0; d < DD; ++d) { float v = src[(size_t)d * SP + s]; ss += v * v; }
    float rinv = 1.0f / fmaxf(sqrtf(ss), 1e-12f);
    __bf16* dst = ofnT + (size_t)b0 * SP * DD + (size_t)s * DD;
    for (int d0 = 0; d0 < DD; d0 += 8) {
        v8bf pack;
#pragma unroll
        for (int i = 0; i < 8; ++i)
            pack[i] = (__bf16)(src[(size_t)(d0 + i) * SP + s] * rinv);
        *(v8bf*)(dst + d0) = pack;
    }
}

// ------------------------------------------------------------------
// 2) inverse column norms of x (over C)
// ------------------------------------------------------------------
__global__ void prep_xinv(const float* __restrict__ xq,
                          const float* __restrict__ xk,
                          float* __restrict__ xinv) {
    int b0 = blockIdx.y;
    int s  = blockIdx.x * 256 + threadIdx.x;
    const float* src = (b0 < BB) ? (xq + (size_t)b0 * CC * SP)
                                 : (xk + (size_t)(b0 - BB) * CC * SP);
    float ss = 0.f;
    for (int c = 0; c < CC; ++c) { float v = src[(size_t)c * SP + s]; ss += v * v; }
    xinv[b0 * SP + s] = 1.0f / fmaxf(sqrtf(ss), 1e-12f);
}

// ------------------------------------------------------------------
// 3) normalize avgpooled_global rows
// ------------------------------------------------------------------
__global__ void prep_agn(const float* __restrict__ agq,
                         const float* __restrict__ agk,
                         float* __restrict__ agn) {
    int i = threadIdx.x;  // 0..127
    const float* src = (i < BB) ? (agq + i * CC) : (agk + (i - BB) * CC);
    float ss = 0.f;
    for (int c = 0; c < CC; ++c) ss += src[c] * src[c];
    float rinv = 1.0f / fmaxf(sqrtf(ss), 1e-12f);
    for (int c = 0; c < CC; ++c) agn[i * CC + c] = src[c] * rinv;
}

// ------------------------------------------------------------------
// 4) BF16 WMMA batched sim + row argmax over t.
//    M = t (argmax dim), N = s, K = d.  Two s-tiles per wave (two resident
//    B fragment sets, 128 VGPRs) + double-buffered A tile (128 VGPRs):
//    tile mt+1's loads issue while tile mt's 16 WMMAs run (two independent
//    accumulator chains).  __launch_bounds__(256,2) -> 512-VGPR budget so
//    nothing spills.  Branchless argmax (cndmask only).
// ------------------------------------------------------------------
__global__ void __launch_bounds__(256, 2)
wmma_argmax(const __bf16* __restrict__ ofnT,
            int* __restrict__ ind) {
    int bd   = blockIdx.y;            // 0..127 : batch-direction task
    int dir  = bd >> 6;
    int b    = bd & 63;
    int wave = threadIdx.x >> 5;
    int lane = threadIdx.x & 31;
    int stp  = blockIdx.x * 8 + wave; // s-tile pair, 0..31
    int st0  = stp * 2;
    int st1  = st0 + 1;
    int aB = (dir == 0) ? (BB + b) : b;  // t-side (argmax dimension)
    int bB = (dir == 0) ? b : (BB + b);  // s-side
    const __bf16* Am = ofnT + (size_t)aB * SP * DD;
    const __bf16* Bm = ofnT + (size_t)bB * SP * DD;
    int  l16 = lane & 15;
    bool hi  = lane >= 16;

    // Two resident B fragment sets (2 x 64 VGPRs)
    v16bf bf0[8], bf1[8];
    {
        const __bf16* brow0 = Bm + (size_t)(st0 * 16 + l16) * DD + (hi ? 16 : 0);
        const __bf16* brow1 = Bm + (size_t)(st1 * 16 + l16) * DD + (hi ? 16 : 0);
#pragma unroll
        for (int kf = 0; kf < 8; ++kf) {
            bf0[kf] = *(const v16bf*)(brow0 + kf * 32);
            bf1[kf] = *(const v16bf*)(brow1 + kf * 32);
        }
    }

    float best0 = -3.4e38f, best1 = -3.4e38f;
    int   bi0 = 0, bi1 = 0;

    const __bf16* abase = Am + (size_t)l16 * DD + (hi ? 16 : 0);

    auto loadA = [&](int mt, v8bf (&lo)[8], v8bf (&hh)[8]) {
        const __bf16* arow = abase + (size_t)mt * (16 * DD);
#pragma unroll
        for (int kf = 0; kf < 8; ++kf) {
            lo[kf] = *(const v8bf*)(arow + kf * 32);
            hh[kf] = *(const v8bf*)(arow + kf * 32 + 8);
        }
    };

    auto compute = [&](int mt, v8bf (&lo)[8], v8bf (&hh)[8]) {
        v8f c0 = {0.f, 0.f, 0.f, 0.f, 0.f, 0.f, 0.f, 0.f};
        v8f c1 = {0.f, 0.f, 0.f, 0.f, 0.f, 0.f, 0.f, 0.f};
#pragma unroll
        for (int kf = 0; kf < 8; ++kf) {
            v16bf a;
#pragma unroll
            for (int i = 0; i < 8; ++i) { a[i] = lo[kf][i]; a[8 + i] = hh[kf][i]; }
            c0 = __builtin_amdgcn_wmma_f32_16x16x32_bf16(
                     false, a, false, bf0[kf], (short)0, c0, false, false);
            c1 = __builtin_amdgcn_wmma_f32_16x16x32_bf16(
                     false, a, false, bf1[kf], (short)0, c1, false, false);
        }
        int tb = mt * 16 + (hi ? 8 : 0);
#pragma unroll
        for (int r = 0; r < 8; ++r) {
            int   t  = tb + r;
            float v0 = c0[r];
            float v1 = c1[r];
            bool g0 = (v0 > best0) | ((v0 == best0) & (t < bi0));
            bool g1 = (v1 > best1) | ((v1 == best1) & (t < bi1));
            best0 = g0 ? v0 : best0;  bi0 = g0 ? t : bi0;
            best1 = g1 ? v1 : best1;  bi1 = g1 ? t : bi1;
        }
    };

    // double-buffered A: prefetch next tile while current tile computes
    v8bf aL0[8], aH0[8], aL1[8], aH1[8];
    loadA(0, aL0, aH0);
    for (int mt = 0; mt < 64; mt += 2) {
        loadA(mt + 1, aL1, aH1);          // in flight during compute(mt)
        compute(mt, aL0, aH0);
        loadA((mt + 2) & 63, aL0, aH0);   // in flight during compute(mt+1)
        compute(mt + 1, aL1, aH1);
    }

    // combine lane L with lane L^16 (same s column, complementary t rows)
    {
        float ov = __shfl_xor(best0, 16, 32);
        int   oi = __shfl_xor(bi0,   16, 32);
        bool g = (ov > best0) | ((ov == best0) & (oi < bi0));
        best0 = g ? ov : best0;  bi0 = g ? oi : bi0;
    }
    {
        float ov = __shfl_xor(best1, 16, 32);
        int   oi = __shfl_xor(bi1,   16, 32);
        bool g = (ov > best1) | ((ov == best1) & (oi < bi1));
        best1 = g ? ov : best1;  bi1 = g ? oi : bi1;
    }
    if (lane < 16) {
        int base = dir * (BB * SP) + b * SP;
        ind[base + st0 * 16 + lane] = bi0;
        ind[base + st1 * 16 + lane] = bi1;
    }
}

// ------------------------------------------------------------------
// 5) dense CE: per row (dir,b,s): l_pos + 63 negatives -> logsumexp.
//    One thread per row; ad (64x128) staged in LDS; deterministic tree sum.
// ------------------------------------------------------------------
__global__ void dense_ce(const float* __restrict__ xq,
                         const float* __restrict__ xk,
                         const float* __restrict__ adq,
                         const float* __restrict__ adk,
                         const float* __restrict__ xinv,
                         const int*   __restrict__ ind,
                         float* __restrict__ partial) {
    __shared__ float ad_sh[BB * CC];   // 32KB
    __shared__ float red[256];
    int tid  = threadIdx.x;
    int row0 = blockIdx.x * 256;       // 256 rows per block, same (dir,b)
    int dir  = row0 >> 16;
    int b    = (row0 >> 10) & 63;
    int s    = (row0 & 1023) + tid;

    const float* adsrc = (dir == 0) ? adk : adq;
    for (int i = tid; i < BB * CC; i += 256) ad_sh[i] = adsrc[i];
    __syncthreads();

    const float* xqm = (dir == 0) ? (xq + (size_t)b * CC * SP) : (xk + (size_t)b * CC * SP);
    const float* xkm = (dir == 0) ? (xk + (size_t)b * CC * SP) : (xq + (size_t)b * CC * SP);
    int qb = (dir == 0) ? b : (BB + b);
    int kb = (dir == 0) ? (BB + b) : b;
    float rinvq = xinv[qb * SP + s];
    int   t     = ind[dir * (BB * SP) + b * SP + s];
    float rinvk = xinv[kb * SP + t];

    float acc[BB];
#pragma unroll
    for (int j = 0; j < BB; ++j) acc[j] = 0.f;
    float accp = 0.f;
    for (int c = 0; c < CC; ++c) {
        float xv = xqm[(size_t)c * SP + s];
        accp += xv * xkm[(size_t)c * SP + t];
#pragma unroll
        for (int j = 0; j < BB; ++j) acc[j] += xv * ad_sh[j * CC + c];
    }
    float pos = accp * rinvq * rinvk * TEMP_INV;
#pragma unroll
    for (int j = 0; j < BB; ++j) acc[j] = acc[j] * rinvq * TEMP_INV;

    // stable two-pass logsumexp over {pos} U {acc[j] : j != b}
    float m = pos;
#pragma unroll
    for (int j = 0; j < BB; ++j) if (j != b) m = fmaxf(m, acc[j]);
    float sum = expf(pos - m);
#pragma unroll
    for (int j = 0; j < BB; ++j) if (j != b) sum += expf(acc[j] - m);
    float contrib = (m + logf(sum)) - pos;

    red[tid] = contrib;
    __syncthreads();
    for (int off = 128; off > 0; off >>= 1) {
        if (tid < off) red[tid] += red[tid + off];
        __syncthreads();
    }
    if (tid == 0) partial[blockIdx.x] = red[0];
}

// ------------------------------------------------------------------
// 6) global CE: 128 threads, one row each; online LSE; shuffle reduce.
// ------------------------------------------------------------------
__global__ void global_loss(const float* __restrict__ agn,
                            float* __restrict__ gsum) {
    int i  = threadIdx.x;         // 0..127
    int pc = (i + BB) & 127;
    const float* ri = agn + i * CC;
    float pos = 0.f;
    for (int c = 0; c < CC; ++c) pos += ri[c] * agn[pc * CC + c];
    pos *= TEMP_INV;
    float m = pos, sum = 1.f;
    for (int j = 0; j < 128; ++j) {
        if (j == i || j == pc) continue;
        float d = 0.f;
        for (int c = 0; c < CC; ++c) d += ri[c] * agn[j * CC + c];
        d *= TEMP_INV;
        if (d > m) { sum = sum * expf(m - d) + 1.f; m = d; }
        else         sum += expf(d - m);
    }
    float contrib = m + logf(sum) - pos;
    for (int off = 16; off > 0; off >>= 1)
        contrib += __shfl_xor(contrib, off, 32);
    __shared__ float wred[4];
    if ((threadIdx.x & 31) == 0) wred[threadIdx.x >> 5] = contrib;
    __syncthreads();
    if (threadIdx.x == 0) gsum[0] = wred[0] + wred[1] + wred[2] + wred[3];
}

// ------------------------------------------------------------------
// 7) finalize (single thread, fully ordered -> deterministic)
// ------------------------------------------------------------------
__global__ void finalize(const float* __restrict__ partial,
                         const float* __restrict__ gsum,
                         const int*   __restrict__ epoch,
                         float* __restrict__ out) {
    if (threadIdx.x == 0 && blockIdx.x == 0) {
        float ds = 0.f;
        for (int i = 0; i < 512; ++i) ds += partial[i];
        float dense = ds / 131072.0f;
        float glob  = gsum[0] / 128.0f;
        out[0] = (epoch[0] > 0) ? (0.5f * glob + 0.5f * dense) : glob;
    }
}

extern "C" void kernel_launch(void* const* d_in, const int* in_sizes, int n_in,
                              void* d_out, int out_size, void* d_ws, size_t ws_size,
                              hipStream_t stream) {
    const float* ofq = (const float*)d_in[0];
    const float* agq = (const float*)d_in[1];
    const float* xq  = (const float*)d_in[2];
    const float* adq = (const float*)d_in[3];
    const float* ofk = (const float*)d_in[4];
    const float* agk = (const float*)d_in[5];
    const float* xk  = (const float*)d_in[6];
    const float* adk = (const float*)d_in[7];
    const int*   ep  = (const int*)d_in[8];
    float* out = (float*)d_out;

    char* ws = (char*)d_ws;
    __bf16* ofnT   = (__bf16*)(ws + OFNT_OFF);
    float*  xinv   = (float*) (ws + XINV_OFF);
    int*    ind    = (int*)   (ws + IND_OFF);
    float*  agn    = (float*) (ws + AGN_OFF);
    float*  dpart  = (float*) (ws + DPART_OFF);
    float*  gsum   = (float*) (ws + GSUM_OFF);

    prep_ofnT   <<<dim3(4, 128), 256, 0, stream>>>(ofq, ofk, ofnT);
    prep_xinv   <<<dim3(4, 128), 256, 0, stream>>>(xq, xk, xinv);
    prep_agn    <<<1, 128, 0, stream>>>(agq, agk, agn);
    wmma_argmax <<<dim3(4, 128), 256, 0, stream>>>(ofnT, ind);
    global_loss <<<1, 128, 0, stream>>>(agn, gsum);
    dense_ce    <<<512, 256, 0, stream>>>(xq, xk, adq, adk, xinv, ind, dpart);
    finalize    <<<1, 64, 0, stream>>>(dpart, gsum, ep, out);
}
